// MHSA_20684562498082
// MI455X (gfx1250) — matrix operand order repaired and checked by reading, not compiled, not run on hardware
//
#include <hip/hip_runtime.h>
#include <hip/hip_bf16.h>
#include <stdint.h>

// MHSA for MI455X / gfx1250 (wave32, WMMA).
// Pipeline: [gemm qkv] -> [rope+split] -> [flash attention] -> [gemm proj]
// Matrix math: v_wmma_f32_16x16x32_bf16 (bf16 operands, fp32 accum).
// Flash attention: double-buffered GLOBAL_LOAD_ASYNC_TO_LDS_B128 (ASYNCcnt)
// overlaps the next K/V tile DMA with the current tile's WMMAs.

typedef __bf16 bf16_t;
typedef __attribute__((ext_vector_type(16))) __bf16 v16bf;
typedef __attribute__((ext_vector_type(8)))  __bf16 v8bf;
typedef __attribute__((ext_vector_type(8)))  float  v8f;

#define USE_ASYNC_LDS 1

// attention LDS row stride: 64 + 8 pad halfs -> 144B rows, conflict-free b128
constexpr int ALD = 72;

#if USE_ASYNC_LDS
// Async DMA: global -> LDS copy from vaddr to vdst, 16B per active lane.
// INST_OFFSET applies to BOTH sides (ISA ch.10.4 async pseudocode).
#define ASYNC_LDS_B128(ldsoff, gaddr, imm)                                   \
  asm volatile("global_load_async_to_lds_b128 %0, %1, off offset:" #imm     \
               :: "v"(ldsoff), "v"(gaddr) : "memory")
__device__ __forceinline__ void wait_asynccnt0() {
  asm volatile("s_wait_asynccnt 0x0" ::: "memory");
}
// Async loads complete in order: counter<=4 means everything older than the
// 4 just-issued prefetch loads has landed.
__device__ __forceinline__ void wait_asynccnt4() {
  asm volatile("s_wait_asynccnt 0x4" ::: "memory");
}
// Flat LDS addressing truncates to addr[31:0] -> wave-relative LDS offset.
__device__ __forceinline__ uint32_t lds_off(const void* p) {
  return (uint32_t)(uintptr_t)p;
}
#endif

__device__ __forceinline__ v8f wmma_bf16(v16bf a, v16bf b, v8f c) {
  // (neg_a, A, neg_b, B, c_mod, C, reuse_a, reuse_b)
  return __builtin_amdgcn_wmma_f32_16x16x32_bf16(false, a, false, b, (short)0, c,
                                                 false, false);
}

// A-operand fragment (16x32 bf16, M rows striped in-lane):
// lanes 0-15: row=lane,    K = {0..7, 16..23}
// lanes 16-31: row=lane-16, K = {8..15, 24..31}
__device__ __forceinline__ v16bf load_a_frag(const bf16_t* p, int row0, int ld,
                                             int kbase) {
  const int lane = threadIdx.x & 31;
  const int r = row0 + (lane & 15);
  const int k = kbase + ((lane >> 4) << 3);
  union { v16bf v; v8bf h[2]; } u;
  u.h[0] = *(const v8bf*)(p + r * ld + k);
  u.h[1] = *(const v8bf*)(p + r * ld + k + 16);
  return u.v;
}

// B-operand fragment (32x16 bf16, presented here as a row-major NxK matrix):
// lanes 0-15: col=lane, K = {0..7, 16..23} ; lanes 16-31: col=lane-16, K = {8..15, 24..31}
__device__ __forceinline__ v16bf load_b_frag(const bf16_t* p, int row0, int ld,
                                             int kbase) {
  const int lane = threadIdx.x & 31;
  const int r = row0 + (lane & 15);
  const int k = kbase + ((lane >> 4) << 3);
  union { v16bf v; v8bf h[2]; } u;
  u.h[0] = *(const v8bf*)(p + r * ld + k);
  u.h[1] = *(const v8bf*)(p + r * ld + k + 16);
  return u.v;
}

// reductions across the 16-lane half that owns one fragment row
__device__ __forceinline__ float half16_max(float v) {
  v = fmaxf(v, __shfl_xor(v, 1, 32));
  v = fmaxf(v, __shfl_xor(v, 2, 32));
  v = fmaxf(v, __shfl_xor(v, 4, 32));
  v = fmaxf(v, __shfl_xor(v, 8, 32));
  return v;
}
__device__ __forceinline__ float half16_sum(float v) {
  v += __shfl_xor(v, 1, 32);
  v += __shfl_xor(v, 2, 32);
  v += __shfl_xor(v, 4, 32);
  v += __shfl_xor(v, 8, 32);
  return v;
}

// ---------------------------------------------------------------------------
// Tiled GEMM: C[M,N] = X[M,K] * W[N,K]^T  (einsum "mk,nk->mn")
// 128x128 C tile / block, BK=32, 8 waves in a 4x2 grid, 2x4 fragments/wave.
// X is fp32 or bf16 (template); W is fp32; both converted to bf16 in LDS
// (conversion must go through VGPRs, so no async DMA here).
// ---------------------------------------------------------------------------
template <typename TIN>
__global__ __launch_bounds__(256) void gemm_xwT(const TIN* __restrict__ X,
                                                const float* __restrict__ W,
                                                float* __restrict__ C,
                                                int M, int N, int K) {
  constexpr int LDT = 40;  // 32 + 8 pad halfs: 80B rows -> conflict-free b128
  __shared__ bf16_t Xs[128 * LDT];
  __shared__ bf16_t Ws[128 * LDT];

  const int tid = threadIdx.x;
  const int lane = tid & 31, w = tid >> 5;
  const int wm = w & 3, wn = w >> 2;  // 4 waves along M, 2 along N
  const int m0 = blockIdx.y * 128, n0 = blockIdx.x * 128;

  // tile fill: 2 threads per row, 16 elements each
  const int tr = tid >> 1;
  const int tk = (tid & 1) * 16;

  v8f acc[2][4];
#pragma unroll
  for (int i = 0; i < 2; ++i)
#pragma unroll
    for (int j = 0; j < 4; ++j) acc[i][j] = {};

  for (int k0 = 0; k0 < K; k0 += 32) {
    if (k0 + 32 < K) {  // prefetch next K-slab (global_prefetch_b8)
      __builtin_prefetch(X + (size_t)(m0 + tr) * K + k0 + 32 + tk, 0, 1);
      __builtin_prefetch(W + (size_t)(n0 + tr) * K + k0 + 32 + tk, 0, 1);
    }
    __syncthreads();
    {  // X tile (convert to bf16 if fp32)
      alignas(16) bf16_t tmp[16];
      const TIN* s = X + (size_t)(m0 + tr) * K + k0 + tk;
#pragma unroll
      for (int i = 0; i < 16; ++i) tmp[i] = (bf16_t)s[i];
      *(v8bf*)(Xs + tr * LDT + tk) = *(const v8bf*)(tmp);
      *(v8bf*)(Xs + tr * LDT + tk + 8) = *(const v8bf*)(tmp + 8);
    }
    {  // W tile (fp32 -> bf16)
      alignas(16) bf16_t tmp[16];
      const float* s = W + (size_t)(n0 + tr) * K + k0 + tk;
#pragma unroll
      for (int i = 0; i < 16; ++i) tmp[i] = (bf16_t)s[i];
      *(v8bf*)(Ws + tr * LDT + tk) = *(const v8bf*)(tmp);
      *(v8bf*)(Ws + tr * LDT + tk + 8) = *(const v8bf*)(tmp + 8);
    }
    __syncthreads();

    v16bf af[2], bfg[4];
#pragma unroll
    for (int mi = 0; mi < 2; ++mi) af[mi] = load_a_frag(Xs, wm * 32 + mi * 16, LDT, 0);
#pragma unroll
    for (int ni = 0; ni < 4; ++ni) bfg[ni] = load_b_frag(Ws, wn * 64 + ni * 16, LDT, 0);
#pragma unroll
    for (int mi = 0; mi < 2; ++mi)
#pragma unroll
      for (int ni = 0; ni < 4; ++ni)
        acc[mi][ni] = wmma_bf16(af[mi], bfg[ni], acc[mi][ni]);
  }

  // C/D layout: VGPR j, lanes 0-15 -> row j, lanes 16-31 -> row j+8; col = lane&15
  const int hf = lane >> 4, cn = lane & 15;
#pragma unroll
  for (int mi = 0; mi < 2; ++mi)
#pragma unroll
    for (int ni = 0; ni < 4; ++ni)
#pragma unroll
      for (int j = 0; j < 8; ++j) {
        const int row = m0 + wm * 32 + mi * 16 + hf * 8 + j;
        const int col = n0 + wn * 64 + ni * 16 + cn;
        C[(size_t)row * N + col] = acc[mi][ni][j];
      }
}

// ---------------------------------------------------------------------------
// RoPE on q,k + head split. Q,K -> (B,H,T,64) bf16 ; V -> (B,H,64,T) bf16
// (V transposed so P·V B-fragments are contiguous LDS rows).
// One thread per rotation pair: idx over B*T*(D/2) = 2*2048*512.
// ---------------------------------------------------------------------------
__global__ __launch_bounds__(256) void rope_split(const float* __restrict__ qkv,
                                                  const float* __restrict__ rope,
                                                  bf16_t* __restrict__ Q,
                                                  bf16_t* __restrict__ Kh,
                                                  bf16_t* __restrict__ Vt) {
  const int idx = blockIdx.x * 256 + threadIdx.x;
  const int i = idx & 511;            // pair index within D/2
  const int t = (idx >> 9) & 2047;    // sequence position
  const int b = idx >> 20;            // batch
  const int h = i >> 5;               // head = (2i)/64
  const int d = (i & 31) * 2;         // within-head dim

  const float c = rope[((size_t)t * 512 + i) * 2 + 0];
  const float s = rope[((size_t)t * 512 + i) * 2 + 1];
  const float* row = qkv + (size_t)(b * 2048 + t) * 3072;

  const float qr = row[2 * i], qi = row[2 * i + 1];
  const float kr = row[1024 + 2 * i], ki = row[1024 + 2 * i + 1];
  const float v0 = row[2048 + 2 * i], v1 = row[2048 + 2 * i + 1];

  const size_t qkb = (((size_t)(b * 16 + h)) * 2048 + t) * 64 + d;
  Q[qkb]      = (bf16_t)(qr * c - qi * s);
  Q[qkb + 1]  = (bf16_t)(qr * s + qi * c);
  Kh[qkb]     = (bf16_t)(kr * c - ki * s);
  Kh[qkb + 1] = (bf16_t)(kr * s + ki * c);

  const size_t vb = (((size_t)(b * 16 + h)) * 64 + d) * 2048 + t;
  Vt[vb]        = (bf16_t)v0;
  Vt[vb + 2048] = (bf16_t)v1;
}

// issue the 4 per-thread async b128 copies for one 64-key K + V^T tile pair
__device__ __forceinline__ void issue_kv_tile(const bf16_t* __restrict__ Kp,
                                              const bf16_t* __restrict__ Vp,
                                              bf16_t* Ksb, bf16_t* Vsb,
                                              int k0, int tid) {
  const int r = tid >> 2, c0 = (tid & 3) * 16;
  const bf16_t* ksrc = Kp + (size_t)(k0 + r) * 64 + c0;
  const bf16_t* vsrc = Vp + (size_t)r * 2048 + k0 + c0;
  bf16_t* kdst = Ksb + r * ALD + c0;
  bf16_t* vdst = Vsb + r * ALD + c0;
#if USE_ASYNC_LDS
  const uint32_t lk = lds_off(kdst), lv = lds_off(vdst);
  const uint64_t gk = (uint64_t)(uintptr_t)ksrc, gv = (uint64_t)(uintptr_t)vsrc;
  ASYNC_LDS_B128(lk, gk, 0);
  ASYNC_LDS_B128(lk, gk, 16);
  ASYNC_LDS_B128(lv, gv, 0);
  ASYNC_LDS_B128(lv, gv, 16);
#else
  ((v8bf*)kdst)[0] = ((const v8bf*)ksrc)[0];
  ((v8bf*)kdst)[1] = ((const v8bf*)ksrc)[1];
  ((v8bf*)vdst)[0] = ((const v8bf*)vsrc)[0];
  ((v8bf*)vdst)[1] = ((const v8bf*)vsrc)[1];
#endif
}

// ---------------------------------------------------------------------------
// Flash attention (single pass, online softmax).
// Block: 128 queries x one (b,h). 8 waves, 16 query rows each.
// 64-key tiles streamed with DOUBLE-BUFFERED async global->LDS DMA: the tile
// kt+1 copy is in flight (ASYNCcnt<=4 outstanding) while tile kt's 16 WMMAs
// and softmax run. Row stats live in registers aligned with the C/D VGPR
// layout; P is re-fragmented through a wave-private LDS region; O += P·V.
// Output merged heads as (B,T,D) bf16.
// ---------------------------------------------------------------------------
__global__ __launch_bounds__(256) void flash_attn(const bf16_t* __restrict__ Q,
                                                  const bf16_t* __restrict__ Kg,
                                                  const bf16_t* __restrict__ Vt,
                                                  bf16_t* __restrict__ AO) {
  __shared__ bf16_t Qs[128 * ALD];     // 128 q rows x 64 dk
  __shared__ bf16_t Ks[2][64 * ALD];   // double-buffered: 64 key rows x 64 dk
  __shared__ bf16_t Vs[2][64 * ALD];   // double-buffered V^T: 64 d x 64 keys
  __shared__ bf16_t Ps[128 * ALD];     // P: 128 q x 64 keys (wave-private rows)

  const int tid = threadIdx.x, lane = tid & 31, w = tid >> 5;
  const int qt = blockIdx.x, bh = blockIdx.y;
  const int b = bh >> 4, h = bh & 15;
  const int q0 = qt * 128;
  const float scale = 0.125f;  // 1/sqrt(64)
  const int hf = lane >> 4, cn = lane & 15;

  const bf16_t* Qp = Q + (size_t)bh * 2048 * 64;
  const bf16_t* Kp = Kg + (size_t)bh * 2048 * 64;
  const bf16_t* Vp = Vt + (size_t)bh * 64 * 2048;

  {  // load Q tile once: 2 threads/row, 64B each (4 x b128 async DMA)
    const int r = tid >> 1, c0 = (tid & 1) * 32;
    const bf16_t* src = Qp + (size_t)(q0 + r) * 64 + c0;
    bf16_t* dst = Qs + r * ALD + c0;
#if USE_ASYNC_LDS
    const uint32_t l = lds_off(dst);
    const uint64_t g = (uint64_t)(uintptr_t)src;
    ASYNC_LDS_B128(l, g, 0);
    ASYNC_LDS_B128(l, g, 16);
    ASYNC_LDS_B128(l, g, 32);
    ASYNC_LDS_B128(l, g, 48);
#else
    ((v8bf*)dst)[0] = ((const v8bf*)src)[0];
    ((v8bf*)dst)[1] = ((const v8bf*)src)[1];
    ((v8bf*)dst)[2] = ((const v8bf*)src)[2];
    ((v8bf*)dst)[3] = ((const v8bf*)src)[3];
#endif
  }
  // prologue: start tile 0 into buffer 0, then drain and publish
  issue_kv_tile(Kp, Vp, Ks[0], Vs[0], 0, tid);
#if USE_ASYNC_LDS
  wait_asynccnt0();
#endif
  __syncthreads();

  v16bf aq[2];
  aq[0] = load_a_frag(Qs, w * 16, ALD, 0);
  aq[1] = load_a_frag(Qs, w * 16, ALD, 32);

  v8f oacc[4];
#pragma unroll
  for (int i = 0; i < 4; ++i) oacc[i] = {};
  float mrow[8], lrow[8];
#pragma unroll
  for (int j = 0; j < 8; ++j) { mrow[j] = -3.0e38f; lrow[j] = 0.0f; }

  for (int kt = 0; kt < 32; ++kt) {
    const int cur = kt & 1;
    // (a) every wave has finished reading buf[cur^1] (consumed in kt-1)
    __syncthreads();
    if (kt < 31) {  // uniform branch: EXEC stays all-1s
      issue_kv_tile(Kp, Vp, Ks[cur ^ 1], Vs[cur ^ 1], (kt + 1) * 64, tid);
#if USE_ASYNC_LDS
      wait_asynccnt4();  // in-order: current tile's 4 older loads have landed
#endif
    } else {
#if USE_ASYNC_LDS
      wait_asynccnt0();
#endif
    }
    // (b) all waves' current-tile data visible in LDS
    __syncthreads();

    const bf16_t* Kc = Ks[cur];
    const bf16_t* Vc = Vs[cur];

    // S = Q K^T : 16 query rows x 64 keys -> 4 fragments, K-dim = dk (2x32)
    v8f sacc[4];
#pragma unroll
    for (int nt = 0; nt < 4; ++nt) {
      sacc[nt] = {};
      sacc[nt] = wmma_bf16(aq[0], load_b_frag(Kc, nt * 16, ALD, 0), sacc[nt]);
      sacc[nt] = wmma_bf16(aq[1], load_b_frag(Kc, nt * 16, ALD, 32), sacc[nt]);
    }

    // online softmax: new row max -> rescale l and O accumulators
#pragma unroll
    for (int j = 0; j < 8; ++j) {
      float tmax = -3.0e38f;
#pragma unroll
      for (int nt = 0; nt < 4; ++nt) tmax = fmaxf(tmax, sacc[nt][j]);
      tmax = half16_max(tmax) * scale;
      const float mnew = fmaxf(mrow[j], tmax);
      const float fac = __expf(mrow[j] - mnew);
      mrow[j] = mnew;
      lrow[j] *= fac;
#pragma unroll
      for (int ot = 0; ot < 4; ++ot) oacc[ot][j] *= fac;
    }

    // P = exp(S*scale - m); accumulate row sums; stage P (bf16) in LDS
#pragma unroll
    for (int nt = 0; nt < 4; ++nt) {
#pragma unroll
      for (int j = 0; j < 8; ++j) {
        const float p = __expf(sacc[nt][j] * scale - mrow[j]);
        lrow[j] += half16_sum(p);
        Ps[(w * 16 + hf * 8 + j) * ALD + nt * 16 + cn] = (bf16_t)p;
      }
    }

    // O += P V   (A from wave-private Ps rows; B from V^T rows; K=keys, 2x32)
    const v16bf ap0 = load_a_frag(Ps, w * 16, ALD, 0);
    const v16bf ap1 = load_a_frag(Ps, w * 16, ALD, 32);
#pragma unroll
    for (int ot = 0; ot < 4; ++ot) {
      oacc[ot] = wmma_bf16(ap0, load_b_frag(Vc, ot * 16, ALD, 0), oacc[ot]);
      oacc[ot] = wmma_bf16(ap1, load_b_frag(Vc, ot * 16, ALD, 32), oacc[ot]);
    }
  }

  // normalize and store merged heads (B,T,D) bf16
#pragma unroll
  for (int ot = 0; ot < 4; ++ot)
#pragma unroll
    for (int j = 0; j < 8; ++j) {
      const int qrow = q0 + w * 16 + hf * 8 + j;
      const int dd = ot * 16 + cn;
      const float val = oacc[ot][j] / lrow[j];
      AO[((size_t)(b * 2048 + qrow)) * 1024 + h * 64 + dd] = (bf16_t)val;
    }
}

// ---------------------------------------------------------------------------
// Host launcher.
// Inputs: x(2,2048,1024) f32, rope(2048,512,2) f32, w_qkv(3072,1024) f32,
//         w_proj(1024,1024) f32. Output: (2,2048,1024) f32.
// Workspace: qkv f32 48MB | Q bf16 8MB | K bf16 8MB | V^T bf16 8MB | AO bf16 8MB
// ---------------------------------------------------------------------------
extern "C" void kernel_launch(void* const* d_in, const int* in_sizes, int n_in,
                              void* d_out, int out_size, void* d_ws, size_t ws_size,
                              hipStream_t stream) {
  (void)in_sizes; (void)n_in; (void)out_size; (void)ws_size;
  const float* x      = (const float*)d_in[0];
  const float* rope   = (const float*)d_in[1];
  const float* w_qkv  = (const float*)d_in[2];
  const float* w_proj = (const float*)d_in[3];
  float* out = (float*)d_out;

  char* ws = (char*)d_ws;
  const size_t QKV_BYTES = (size_t)4096 * 3072 * sizeof(float);  // 50331648
  const size_t HB = (size_t)2 * 16 * 2048 * 64 * sizeof(bf16_t); // 8388608
  float*  qkv = (float*)ws;
  bf16_t* Qw  = (bf16_t*)(ws + QKV_BYTES);
  bf16_t* Kw  = (bf16_t*)(ws + QKV_BYTES + HB);
  bf16_t* Vtw = (bf16_t*)(ws + QKV_BYTES + 2 * HB);
  bf16_t* AOw = (bf16_t*)(ws + QKV_BYTES + 3 * HB);

  // 1) qkv = x @ w_qkv^T : (4096,1024) x (3072,1024)^T
  gemm_xwT<float><<<dim3(24, 32), 256, 0, stream>>>(x, w_qkv, qkv, 4096, 3072, 1024);
  // 2) rope + head split (+ V transpose)
  rope_split<<<8192, 256, 0, stream>>>(qkv, rope, Qw, Kw, Vtw);
  // 3) flash attention over (qtile, b*h)
  flash_attn<<<dim3(16, 32), 256, 0, stream>>>(Qw, Kw, Vtw, AOw);
  // 4) out = attn @ w_proj^T : (4096,1024) x (1024,1024)^T
  gemm_xwT<bf16_t><<<dim3(8, 32), 256, 0, stream>>>(AOw, w_proj, out, 4096, 1024, 1024);
}
